// GatClassification_80479097192672
// MI455X (gfx1250) — matrix operand order repaired and checked by reading, not compiled, not run on hardware
//
#include <hip/hip_runtime.h>
#include <stdint.h>

#define N_NODES 20000
#define T_STEPS 8
#define E_EDGES 320000
#define NFEAT   512
#define HEADS   4
#define CCH     128
#define NHID    128
#define NCLASS  2
#define SLOPE   0.2f
#define ETOT    (E_EDGES + N_NODES)

typedef unsigned short ushort_t;
typedef __attribute__((ext_vector_type(16))) __bf16 v16bf;
typedef __attribute__((ext_vector_type(8)))  float  v8f;

union FragU { uint4 u[2]; v16bf bf; };

__device__ __forceinline__ float leaky_f(float x) { return x >= 0.f ? x : SLOPE * x; }

__device__ __forceinline__ ushort_t f2bf(float x) {
    unsigned int u = __float_as_uint(x);
    u += 0x7FFFu + ((u >> 16) & 1u);     // round-to-nearest-even
    return (ushort_t)(u >> 16);
}

// ---------------------------------------------------------------- elementwise
__global__ __launch_bounds__(256) void cvt_bf16_kernel(const float* __restrict__ src,
                                                       ushort_t* __restrict__ dst, int n) {
    int i = blockIdx.x * blockDim.x + threadIdx.x;
    if (i < n) dst[i] = f2bf(src[i]);
}

__global__ __launch_bounds__(256) void fill_kernel(float* __restrict__ p, float v, int n) {
    int i = blockIdx.x * blockDim.x + threadIdx.x;
    if (i < n) p[i] = v;
}

// --------------------------------------------------------- pack W into B-frag
// B tile = 32(K) x 16(N), 16 bf16 per lane: lane 0-15 -> N=lane, K=0..15;
// lane 16-31 -> N=lane-16, K=16..31. ushort j within lane <-> K = kb + j.
__global__ void pack_b_kernel(const float* __restrict__ W, ushort_t* __restrict__ Wp,
                              int K, int Nout) {
    int tile = blockIdx.x;                 // tile = nt*(K/32) + kt
    int ktiles = K >> 5;
    int nt = tile / ktiles, kt = tile % ktiles;
    int lane = threadIdx.x;
    int ncol = nt * 16 + (lane & 15);
    int kb = (lane < 16) ? 0 : 16;
    ushort_t* out = Wp + ((size_t)tile * 32 + lane) * 16;
#pragma unroll
    for (int j = 0; j < 16; ++j)
        out[j] = f2bf(W[(size_t)(kt * 32 + kb + j) * Nout + ncol]);
}

// ------------------------------------------------------------- WMMA bf16 GEMM
// C[M,NOUT] = A[M,K](bf16) * B(packed bf16) (+bias).
// Block = 8 waves; each wave owns a 32x64 output tile (2 A-frags x 4 B-frags
// -> 8 WMMA accumulators), so every B fragment feeds two WMMAs (32 flop/B
// from L2). The block's A rows (32 KB for both instantiations) are staged
// into LDS with CDNA5 async loads (ASYNCcnt) and read back with ds_load_b128.
template <int K, int NOUT>
__global__ __launch_bounds__(256) void wmma_gemm_lds_kernel(
        const ushort_t* __restrict__ Abf, const ushort_t* __restrict__ Bp,
        float* __restrict__ Cout, const float* __restrict__ bias, int M) {
    constexpr int NGROUPS = NOUT / 64;          // 64-column groups
    constexpr int KTILES  = K / 32;
    constexpr int ROWS    = 32 * (8 / NGROUPS); // A rows staged per block
    static_assert(ROWS * K == 16384, "LDS tile must be 32KB");
    __shared__ alignas(16) ushort_t tileA[ROWS * K];

    const int tid   = threadIdx.x;
    const int lane  = tid & 31;
    const int wv    = tid >> 5;
    const int tile0 = blockIdx.x * 8;                 // first wave-tile in block
    const int row_base = (tile0 / NGROUPS) * 32;      // first A row staged

    // ---- async stage: 32 KB of A rows -> LDS (128 B per thread) ----
    {
        unsigned lbase = (unsigned)(uintptr_t)(&tileA[0]) + (unsigned)tid * 16u;
        unsigned long long gbase =
            (unsigned long long)(uintptr_t)(Abf + (size_t)row_base * K) +
            (unsigned long long)tid * 16u;
#pragma unroll
        for (int j = 0; j < 8; ++j) {
            unsigned l = lbase + (unsigned)j * 4096u;
            unsigned long long g = gbase + (unsigned long long)j * 4096u;
            asm volatile("global_load_async_to_lds_b128 %0, %1, off"
                         :: "v"(l), "v"(g) : "memory");
        }
        asm volatile("s_wait_asynccnt 0x0" ::: "memory");
    }
    __syncthreads();

    const int wtile = tile0 + wv;
    const int tiles_m = (M + 31) >> 5;                // 32-row tiles
    if (wtile >= tiles_m * NGROUPS) return;           // idle waves (after barrier)
    const int tm = wtile / NGROUPS;
    const int tg = wtile % NGROUPS;
    const int half = lane >> 4;                       // A K-chunk / C row-half
    const int lrow = (tm * 32 - row_base) + (lane & 15);

    const ushort_t* arow0 = tileA + lrow * K + half * 16;       // rows 0..15
    const ushort_t* arow1 = arow0 + 16 * K;                     // rows 16..31
    const ushort_t* bcol  = Bp + (size_t)(tg * 4) * KTILES * 512 + (size_t)lane * 16;
    constexpr size_t BSTRIDE = (size_t)KTILES * 512;  // ushorts per 16-col block

    v8f acc0[4] = {};
    v8f acc1[4] = {};
#pragma unroll
    for (int kt = 0; kt < KTILES; ++kt) {
        FragU a0, a1;
        a0.u[0] = *(const uint4*)(arow0 + kt * 32);      // ds_load_b128
        a0.u[1] = *(const uint4*)(arow0 + kt * 32 + 8);
        a1.u[0] = *(const uint4*)(arow1 + kt * 32);
        a1.u[1] = *(const uint4*)(arow1 + kt * 32 + 8);
#pragma unroll
        for (int nn = 0; nn < 4; ++nn) {
            FragU b;
            b.u[0] = *(const uint4*)(bcol + nn * BSTRIDE + kt * 512);
            b.u[1] = *(const uint4*)(bcol + nn * BSTRIDE + kt * 512 + 8);
            acc0[nn] = __builtin_amdgcn_wmma_f32_16x16x32_bf16(
                false, a0.bf, false, b.bf, (short)0, acc0[nn], false, false);
            acc1[nn] = __builtin_amdgcn_wmma_f32_16x16x32_bf16(
                false, a1.bf, false, b.bf, (short)0, acc1[nn], false, false);
        }
    }

    const int rb0 = tm * 32 + half * 8;               // C: VGPR r -> row r + half*8
    const int rb1 = rb0 + 16;
    const int colbase = tg * 64 + (lane & 15);
#pragma unroll
    for (int nn = 0; nn < 4; ++nn) {
        int col = colbase + nn * 16;
        float bv = bias ? bias[col] : 0.f;
#pragma unroll
        for (int r = 0; r < 8; ++r) {
            Cout[(size_t)(rb0 + r) * NOUT + col] = acc0[nn][r] + bv;
            Cout[(size_t)(rb1 + r) * NOUT + col] = acc1[nn][r] + bv;
        }
    }
}

// -------------------------------------------------- per-(node,head) attn dots
__global__ __launch_bounds__(256) void att_kernel(const float* __restrict__ h,
                                                  const float* __restrict__ att_s,
                                                  const float* __restrict__ att_d,
                                                  float* __restrict__ a_s,
                                                  float* __restrict__ a_d) {
    int lane = threadIdx.x & 31;
    int wave = (int)((blockIdx.x * blockDim.x + threadIdx.x) >> 5);
    if (wave >= N_NODES * HEADS) return;
    int n = wave >> 2, hh = wave & 3;
    float4 hv = *(const float4*)(h + (size_t)n * 512 + hh * 128 + lane * 4);
    float4 as = *(const float4*)(att_s + hh * 128 + lane * 4);
    float4 ad = *(const float4*)(att_d + hh * 128 + lane * 4);
    float s = hv.x * as.x + hv.y * as.y + hv.z * as.z + hv.w * as.w;
    float d = hv.x * ad.x + hv.y * ad.y + hv.z * ad.z + hv.w * ad.w;
    for (int off = 16; off; off >>= 1) {
        s += __shfl_down(s, off);
        d += __shfl_down(d, off);
    }
    if (lane == 0) { a_s[n * 4 + hh] = s; a_d[n * 4 + hh] = d; }
}

// --------------------------------------------------------- edge passes 1..3
__global__ __launch_bounds__(256) void edge_max_kernel(const int* __restrict__ gsrc,
                                                       const int* __restrict__ gdst,
                                                       const float* __restrict__ a_s,
                                                       const float* __restrict__ a_d,
                                                       float* __restrict__ e,
                                                       float* __restrict__ m) {
    int i = blockIdx.x * blockDim.x + threadIdx.x;
    if (i >= ETOT) return;
    int s = (i < E_EDGES) ? gsrc[i] : (i - E_EDGES);
    int d = (i < E_EDGES) ? gdst[i] : (i - E_EDGES);
#pragma unroll
    for (int hh = 0; hh < HEADS; ++hh) {
        float ee = leaky_f(a_s[s * 4 + hh] + a_d[d * 4 + hh]);
        e[(size_t)i * 4 + hh] = ee;
        __hip_atomic_fetch_max(&m[d * 4 + hh], ee, __ATOMIC_RELAXED, __HIP_MEMORY_SCOPE_AGENT);
    }
}

__global__ __launch_bounds__(256) void edge_exp_kernel(const int* __restrict__ gdst,
                                                       float* __restrict__ e,
                                                       const float* __restrict__ m,
                                                       float* __restrict__ denom) {
    int i = blockIdx.x * blockDim.x + threadIdx.x;
    if (i >= ETOT) return;
    int d = (i < E_EDGES) ? gdst[i] : (i - E_EDGES);
#pragma unroll
    for (int hh = 0; hh < HEADS; ++hh) {
        float ex = __expf(e[(size_t)i * 4 + hh] - m[d * 4 + hh]);
        e[(size_t)i * 4 + hh] = ex;
        __hip_atomic_fetch_add(&denom[d * 4 + hh], ex, __ATOMIC_RELAXED, __HIP_MEMORY_SCOPE_AGENT);
    }
}

// one wave per edge: lane covers 16 channels (single head each, 128/8=16)
__global__ __launch_bounds__(256) void edge_scatter_kernel(const int* __restrict__ gsrc,
                                                           const int* __restrict__ gdst,
                                                           const float* __restrict__ e,
                                                           const float* __restrict__ denom,
                                                           const float* __restrict__ h,
                                                           float* __restrict__ accum) {
    int lane = threadIdx.x & 31;
    int i = (int)((blockIdx.x * blockDim.x + threadIdx.x) >> 5);
    if (i >= ETOT) return;
    int s = (i < E_EDGES) ? gsrc[i] : (i - E_EDGES);
    int d = (i < E_EDGES) ? gdst[i] : (i - E_EDGES);
    int hh = lane >> 3;
    float alpha = e[(size_t)i * 4 + hh] / (denom[d * 4 + hh] + 1e-16f);
    int c0 = lane * 16;
    const float* hp = h + (size_t)s * 512 + c0;
    float* ap = accum + (size_t)d * 512 + c0;
#pragma unroll
    for (int j = 0; j < 4; ++j) {
        float4 v = *(const float4*)(hp + j * 4);
        __hip_atomic_fetch_add(ap + j * 4 + 0, v.x * alpha, __ATOMIC_RELAXED, __HIP_MEMORY_SCOPE_AGENT);
        __hip_atomic_fetch_add(ap + j * 4 + 1, v.y * alpha, __ATOMIC_RELAXED, __HIP_MEMORY_SCOPE_AGENT);
        __hip_atomic_fetch_add(ap + j * 4 + 2, v.z * alpha, __ATOMIC_RELAXED, __HIP_MEMORY_SCOPE_AGENT);
        __hip_atomic_fetch_add(ap + j * 4 + 3, v.w * alpha, __ATOMIC_RELAXED, __HIP_MEMORY_SCOPE_AGENT);
    }
}

// ------------------------------------------- head mean + bias + leaky -> bf16
__global__ __launch_bounds__(256) void head_mean_kernel(const float* __restrict__ accum,
                                                        const float* __restrict__ b_gat,
                                                        ushort_t* __restrict__ ybf) {
    int i = blockIdx.x * blockDim.x + threadIdx.x;
    if (i >= N_NODES * CCH) return;
    int n = i >> 7, c = i & 127;
    const float* a = accum + (size_t)n * 512 + c;
    float v = (a[0] + a[128] + a[256] + a[384]) * 0.25f + b_gat[c];
    ybf[i] = f2bf(leaky_f(v));
}

// ------------------------- leaky + [1024]x[1024,2] dot + log_softmax per node
__global__ __launch_bounds__(256) void final_kernel(const float* __restrict__ out2,
                                                    const float* __restrict__ W2,
                                                    const float* __restrict__ b2,
                                                    float* __restrict__ out) {
    int lane = threadIdx.x & 31;
    int n = (int)((blockIdx.x * blockDim.x + threadIdx.x) >> 5);
    if (n >= N_NODES) return;
    const float* row = out2 + (size_t)n * (T_STEPS * NHID);
    float s0 = 0.f, s1 = 0.f;
    for (int j = lane; j < T_STEPS * NHID; j += 32) {
        float v = leaky_f(row[j]);
        s0 += v * W2[j * 2 + 0];
        s1 += v * W2[j * 2 + 1];
    }
    for (int off = 16; off; off >>= 1) {
        s0 += __shfl_down(s0, off);
        s1 += __shfl_down(s1, off);
    }
    if (lane == 0) {
        float v0 = s0 + b2[0], v1 = s1 + b2[1];
        float mx = fmaxf(v0, v1);
        float lse = mx + __logf(__expf(v0 - mx) + __expf(v1 - mx));
        out[n * 2 + 0] = v0 - lse;
        out[n * 2 + 1] = v1 - lse;
    }
}

// ============================================================================
extern "C" void kernel_launch(void* const* d_in, const int* in_sizes, int n_in,
                              void* d_out, int out_size, void* d_ws, size_t ws_size,
                              hipStream_t stream) {
    (void)in_sizes; (void)n_in; (void)out_size; (void)ws_size;
    const float* fts   = (const float*)d_in[0];
    const int*   graph = (const int*)d_in[1];
    const float* W_gat = (const float*)d_in[3];
    const float* att_s = (const float*)d_in[4];
    const float* att_d = (const float*)d_in[5];
    const float* b_gat = (const float*)d_in[6];
    const float* W1    = (const float*)d_in[7];
    const float* b1    = (const float*)d_in[8];
    const float* W2    = (const float*)d_in[9];
    const float* b2    = (const float*)d_in[10];
    float* out = (float*)d_out;

    char* ws = (char*)d_ws;
    auto alloc = [&](size_t bytes) -> char* {
        char* p = ws;
        ws += (bytes + 255) & ~(size_t)255;
        return p;
    };
    ushort_t* xbf   = (ushort_t*)alloc((size_t)N_NODES * NFEAT * 2);
    ushort_t* Wgp   = (ushort_t*)alloc((size_t)512 * 512 * 2);
    ushort_t* W1p   = (ushort_t*)alloc((size_t)128 * 128 * 2);
    float*    h     = (float*)alloc((size_t)N_NODES * 512 * 4);
    float*    a_s   = (float*)alloc((size_t)N_NODES * 4 * 4);
    float*    a_d   = (float*)alloc((size_t)N_NODES * 4 * 4);
    float*    e     = (float*)alloc((size_t)ETOT * 4 * 4);
    float*    m     = (float*)alloc((size_t)N_NODES * 4 * 4);
    float*    den   = (float*)alloc((size_t)N_NODES * 4 * 4);
    float*    accum = (float*)alloc((size_t)N_NODES * 512 * 4);
    ushort_t* ybf   = (ushort_t*)alloc((size_t)N_NODES * 128 * 2);
    float*    out2  = (float*)alloc((size_t)T_STEPS * N_NODES * 128 * 4);

    // one-time weight packing into WMMA B-fragment layout
    pack_b_kernel<<<(512 / 16) * (512 / 32), 32, 0, stream>>>(W_gat, Wgp, 512, 512);
    pack_b_kernel<<<(128 / 16) * (128 / 32), 32, 0, stream>>>(W1, W1p, 128, 128);

    for (int t = 0; t < T_STEPS; ++t) {
        const float* x   = fts + (size_t)t * N_NODES * NFEAT;
        const int* gsrc  = graph + (size_t)t * 2 * E_EDGES;
        const int* gdst  = gsrc + E_EDGES;

        cvt_bf16_kernel<<<(N_NODES * NFEAT + 255) / 256, 256, 0, stream>>>(x, xbf, N_NODES * NFEAT);

        // h = x @ W_gat   [20000,512]x[512,512]  (8 waves/block, 32 rows staged)
        {
            int blocks = ((N_NODES / 32) * (512 / 64) + 7) / 8;   // 625
            wmma_gemm_lds_kernel<512, 512><<<blocks, 256, 0, stream>>>(
                xbf, Wgp, h, nullptr, N_NODES);
        }
        att_kernel<<<(N_NODES * HEADS + 7) / 8, 256, 0, stream>>>(h, att_s, att_d, a_s, a_d);

        fill_kernel<<<(N_NODES * 4 + 255) / 256, 256, 0, stream>>>(m, -3.0e38f, N_NODES * 4);
        fill_kernel<<<(N_NODES * 4 + 255) / 256, 256, 0, stream>>>(den, 0.f, N_NODES * 4);
        fill_kernel<<<(N_NODES * 512 + 255) / 256, 256, 0, stream>>>(accum, 0.f, N_NODES * 512);

        edge_max_kernel<<<(ETOT + 255) / 256, 256, 0, stream>>>(gsrc, gdst, a_s, a_d, e, m);
        edge_exp_kernel<<<(ETOT + 255) / 256, 256, 0, stream>>>(gdst, e, m, den);
        edge_scatter_kernel<<<(ETOT + 7) / 8, 256, 0, stream>>>(gsrc, gdst, e, den, h, accum);

        head_mean_kernel<<<(N_NODES * CCH + 255) / 256, 256, 0, stream>>>(accum, b_gat, ybf);

        // z_t = y_t @ W1 + b1, written at [t,N,128] so the reshape is free
        {
            int blocks = ((N_NODES / 32) * (128 / 64) + 7) / 8;   // 157 (last partial)
            wmma_gemm_lds_kernel<128, 128><<<blocks, 256, 0, stream>>>(
                ybf, W1p, out2 + (size_t)t * N_NODES * 128, b1, N_NODES);
        }
    }

    final_kernel<<<(N_NODES + 7) / 8, 256, 0, stream>>>(out2, W2, b2, out);
}
